// DiVeQ_78426102825288
// MI455X (gfx1250) — matrix operand order repaired and checked by reading, not compiled, not run on hardware
//
#include <hip/hip_runtime.h>
#include <stdint.h>

// ---------------- problem constants (match reference) ----------------
#define D_DIM  256      // embedding dim (GEMM K)
#define K_CB   8192     // codebook entries (GEMM N)
#define N_ROWS 16384    // flattened z rows (GEMM M)

constexpr float kEps = 1e-8f;
constexpr float kLossScale = 1.25f / ((float)N_ROWS * (float)D_DIM); // codebook + 0.25*commitment, same MSE

// ---------------- CDNA5 WMMA types ----------------
typedef __attribute__((ext_vector_type(16))) __bf16 v16bf;
typedef __attribute__((ext_vector_type(8)))  float  v8f;

// 8-dword fragment (== v16bf == 32 bytes). dword j -> VGPR j of the WMMA operand.
union Frag32 {
  uint32_t u[8];
  uint4    q[2];
  v16bf    v;
};

__device__ __forceinline__ v8f wmma_bf16(v16bf a, v16bf b, v8f c) {
  // v_wmma_f32_16x16x32_bf16 : (neg_a, A, neg_b, B, c_mod, C, reuse_a, reuse_b)
  return __builtin_amdgcn_wmma_f32_16x16x32_bf16(false, a, false, b, (short)0, c, false, false);
}

// ---------------- bf16 split helpers (RNE) ----------------
__device__ __forceinline__ uint16_t f32_to_bf16_rne(float f) {
  uint32_t u = __float_as_uint(f);
  u += 0x7FFFu + ((u >> 16) & 1u);
  return (uint16_t)(u >> 16);
}
__device__ __forceinline__ float bf16_to_f32(uint16_t h) {
  return __uint_as_float(((uint32_t)h) << 16);
}

// ---------------- kernel 1: codebook -> (bf16 hi, bf16 lo, ||c||^2) ----------------
__global__ void __launch_bounds__(256) diveq_prep_codebook(
    const float* __restrict__ cb, uint16_t* __restrict__ c_hi,
    uint16_t* __restrict__ c_lo, float* __restrict__ cnorm) {
  __shared__ float red[256];
  const int k = blockIdx.x;
  const int t = threadIdx.x;
  const float x = cb[k * D_DIM + t];
  const uint16_t h = f32_to_bf16_rne(x);
  c_hi[k * D_DIM + t] = h;
  c_lo[k * D_DIM + t] = f32_to_bf16_rne(x - bf16_to_f32(h));
  red[t] = x * x;
  __syncthreads();
  for (int s = 128; s > 0; s >>= 1) {
    if (t < s) red[t] += red[t + s];
    __syncthreads();
  }
  if (t == 0) cnorm[k] = red[0];
}

// ---------------- kernel 2: z -> (bf16 hi, bf16 lo) ----------------
__global__ void __launch_bounds__(256) diveq_prep_z(
    const float* __restrict__ z, uint16_t* __restrict__ z_hi,
    uint16_t* __restrict__ z_lo) {
  const int i = blockIdx.x * 256 + threadIdx.x;
  const float x = z[i];
  const uint16_t h = f32_to_bf16_rne(x);
  z_hi[i] = h;
  z_lo[i] = f32_to_bf16_rne(x - bf16_to_f32(h));
}

// ---------------- kernel 3: fused bf16x3 WMMA GEMM + running argmin ----------------
// Wave tile: 32 M (2 subtiles of 16) x 16 N per chunk. A resident in VGPRs.
// argmin over k of (||c_k||^2 - 2 z.c_k)  (||z||^2 constant per row -> dropped)
#define M_WAVE 32
#define WAVES  8
#define M_WG   (M_WAVE * WAVES)   // 256 rows per workgroup

__global__ void __launch_bounds__(256, 1) diveq_argmin_wmma(
    const uint16_t* __restrict__ z_hi, const uint16_t* __restrict__ z_lo,
    const uint16_t* __restrict__ c_hi, const uint16_t* __restrict__ c_lo,
    const float* __restrict__ cnorm, int* __restrict__ out_idx) {
  const int wave = threadIdx.x >> 5;
  const int lane = threadIdx.x & 31;
  const int n16  = lane & 15;       // column-within-tile / row-M for A
  const int half = lane >> 4;       // lane half selects K sub-block
  const int rowBase = blockIdx.x * M_WG + wave * M_WAVE;

  // ---- preload resident A fragments: [subtile][kblock], hi & lo ----
  Frag32 aHi[2][8], aLo[2][8];
#pragma unroll
  for (int s = 0; s < 2; ++s) {
    const int row = rowBase + s * 16 + n16;           // A: lane (mod 16) carries M
    const uint16_t* zh = z_hi + (size_t)row * D_DIM;
    const uint16_t* zl = z_lo + (size_t)row * D_DIM;
#pragma unroll
    for (int kb = 0; kb < 8; ++kb) {
      const int e0 = kb * 32 + 8 * half;              // A layout: K = 8h + {0..7}, then +16
      aHi[s][kb].q[0] = *(const uint4*)(zh + e0);
      aHi[s][kb].q[1] = *(const uint4*)(zh + e0 + 16);
      aLo[s][kb].q[0] = *(const uint4*)(zl + e0);
      aLo[s][kb].q[1] = *(const uint4*)(zl + e0 + 16);
    }
  }

  float minv[2][8];
  int   mini[2][8];
#pragma unroll
  for (int s = 0; s < 2; ++s)
#pragma unroll
    for (int v = 0; v < 8; ++v) { minv[s][v] = 3.4e38f; mini[s][v] = 0; }

  for (int chunk = 0; chunk < K_CB / 16; ++chunk) {
    const int col = chunk * 16 + n16;                 // B: lane (mod 16) carries N
    const float cn = cnorm[col];
    const uint16_t* ch = c_hi + (size_t)col * D_DIM + 8 * half; // B layout: K = 8h + {0..7}, then +16
    const uint16_t* cl = c_lo + (size_t)col * D_DIM + 8 * half;
    // prefetch next chunk's codebook rows (global_prefetch_b8)
    __builtin_prefetch(ch + 16 * D_DIM, 0, 0);
    __builtin_prefetch(cl + 16 * D_DIM, 0, 0);

    v8f acc0 = {0.f, 0.f, 0.f, 0.f, 0.f, 0.f, 0.f, 0.f};
    v8f acc1 = {0.f, 0.f, 0.f, 0.f, 0.f, 0.f, 0.f, 0.f};
#pragma unroll
    for (int kb = 0; kb < 8; ++kb) {
      Frag32 bHi, bLo;
      bHi.q[0] = *(const uint4*)(ch + kb * 32);
      bHi.q[1] = *(const uint4*)(ch + kb * 32 + 16);
      bLo.q[0] = *(const uint4*)(cl + kb * 32);
      bLo.q[1] = *(const uint4*)(cl + kb * 32 + 16);
      // bf16x3 compensated product, f32 accumulate
      acc0 = wmma_bf16(aHi[0][kb].v, bHi.v, acc0);
      acc0 = wmma_bf16(aHi[0][kb].v, bLo.v, acc0);
      acc0 = wmma_bf16(aLo[0][kb].v, bHi.v, acc0);
      acc1 = wmma_bf16(aHi[1][kb].v, bHi.v, acc1);
      acc1 = wmma_bf16(aHi[1][kb].v, bLo.v, acc1);
      acc1 = wmma_bf16(aLo[1][kb].v, bHi.v, acc1);
    }
    // C layout: VGPR v -> M = v + 8*half, N = lane%16. Update running min.
#pragma unroll
    for (int v = 0; v < 8; ++v) {
      const float d0 = cn - 2.f * acc0[v];
      if (d0 < minv[0][v]) { minv[0][v] = d0; mini[0][v] = col; }
      const float d1 = cn - 2.f * acc1[v];
      if (d1 < minv[1][v]) { minv[1][v] = d1; mini[1][v] = col; }
    }
  }

  // ---- cross-lane argmin (butterfly within each 16-lane half; ties -> smaller idx) ----
#pragma unroll
  for (int s = 0; s < 2; ++s) {
#pragma unroll
    for (int v = 0; v < 8; ++v) {
      float mv = minv[s][v];
      int   mi = mini[s][v];
#pragma unroll
      for (int m = 1; m <= 8; m <<= 1) {
        const float ov = __shfl_xor(mv, m, 32);
        const int   oi = __shfl_xor(mi, m, 32);
        if (ov < mv || (ov == mv && oi < mi)) { mv = ov; mi = oi; }
      }
      if (n16 == 0) out_idx[rowBase + s * 16 + 8 * half + v] = mi;
    }
  }
}

// ---------------- kernel 4: gather + z_q + per-row squared error ----------------
__global__ void __launch_bounds__(256) diveq_epilogue(
    const float* __restrict__ z, const float* __restrict__ cb,
    const int* __restrict__ idx, float* __restrict__ zq,
    float* __restrict__ idx_out_f, float* __restrict__ row_sums) {
  __shared__ float red[256];
  __shared__ float s_scale;
  const int row = blockIdx.x;
  const int t = threadIdx.x;
  const int k = idx[row];
  const float zf = z[(size_t)row * D_DIM + t];
  const float cf = cb[(size_t)k * D_DIM + t];
  const float d = cf - zf;
  red[t] = d * d;
  __syncthreads();
  for (int s = 128; s > 0; s >>= 1) {
    if (t < s) red[t] += red[t + s];
    __syncthreads();
  }
  if (t == 0) {
    const float ss = red[0];
    row_sums[row] = ss;
    const float mag = sqrtf(ss);
    s_scale = mag / (mag + kEps);       // z_q = z + d * mag/(mag+eps)
    idx_out_f[row] = (float)k;
  }
  __syncthreads();
  zq[(size_t)row * D_DIM + t] = zf + d * s_scale;
}

// ---------------- kernel 5: deterministic loss reduction ----------------
__global__ void __launch_bounds__(256) diveq_loss(
    const float* __restrict__ row_sums, float* __restrict__ loss) {
  __shared__ float red[256];
  const int t = threadIdx.x;
  float s = 0.f;
  for (int r = t; r < N_ROWS; r += 256) s += row_sums[r];
  red[t] = s;
  __syncthreads();
  for (int st = 128; st > 0; st >>= 1) {
    if (t < st) red[t] += red[t + st];
    __syncthreads();
  }
  if (t == 0) *loss = red[0] * kLossScale;
}

// ---------------- host launcher ----------------
extern "C" void kernel_launch(void* const* d_in, const int* in_sizes, int n_in,
                              void* d_out, int out_size, void* d_ws, size_t ws_size,
                              hipStream_t stream) {
  (void)in_sizes; (void)n_in; (void)out_size; (void)ws_size;
  const float* z  = (const float*)d_in[0];   // [4,4096,256] f32
  const float* cb = (const float*)d_in[1];   // [8192,256]   f32

  float* out   = (float*)d_out;
  float* zq    = out;                                   // [N_ROWS*D_DIM]
  float* loss  = out + (size_t)N_ROWS * D_DIM;          // [1]
  float* idx_f = loss + 1;                              // [N_ROWS]

  char* ws = (char*)d_ws;
  float*    cnorm = (float*)ws;     ws += (size_t)K_CB * sizeof(float);
  uint16_t* c_hi  = (uint16_t*)ws;  ws += (size_t)K_CB * D_DIM * 2;
  uint16_t* c_lo  = (uint16_t*)ws;  ws += (size_t)K_CB * D_DIM * 2;
  uint16_t* zhi   = (uint16_t*)ws;  ws += (size_t)N_ROWS * D_DIM * 2;
  uint16_t* zlo   = (uint16_t*)ws;  ws += (size_t)N_ROWS * D_DIM * 2;
  int*      amin  = (int*)ws;       ws += (size_t)N_ROWS * sizeof(int);
  float*    rsum  = (float*)ws;

  diveq_prep_codebook<<<K_CB, 256, 0, stream>>>(cb, c_hi, c_lo, cnorm);
  diveq_prep_z<<<(N_ROWS * D_DIM) / 256, 256, 0, stream>>>(z, zhi, zlo);
  diveq_argmin_wmma<<<N_ROWS / M_WG, 256, 0, stream>>>(zhi, zlo, c_hi, c_lo, cnorm, amin);
  diveq_epilogue<<<N_ROWS, 256, 0, stream>>>(z, cb, amin, zq, idx_f, rsum);
  diveq_loss<<<1, 256, 0, stream>>>(rsum, loss);
}
